// MultiHeadAttention_21912923144687
// MI455X (gfx1250) — matrix operand order repaired and checked by reading, not compiled, not run on hardware
//
#include <hip/hip_runtime.h>
#include <hip/hip_bf16.h>

// MHA forward for B=2, S=2048, D=1024, H=16, Dh=64 on gfx1250 (MI455X).
// All matmuls in bf16 WMMA (v_wmma_f32_16x16x32_bf16), f32 accumulation.
// Straight global->LDS copies use GLOBAL_LOAD_ASYNC_TO_LDS_B128 (ASYNCcnt),
// next-chunk streams are warmed with global_prefetch_b8.

#define D_MODEL   1024
#define NUM_HEADS 16
#define D_H       64
#define SEQ       2048
#define BATCH     2

typedef __bf16 bf16_t;
typedef __attribute__((ext_vector_type(16))) __bf16 v16bf;
typedef __attribute__((ext_vector_type(8)))  float  v8f;

union FragB16 { unsigned u[8]; v16bf v; };

// A-matrix (16x32 bf16) K index for VGPR v, lane-half h:
//   lanes 0-15: V0..V3 -> K 0..7, V4..V7 -> K 16..23 ; lanes 16-31: +8
__device__ __forceinline__ int kmapA(int v, int half) {
    return 2 * v + (v >= 4 ? 8 : 0) + half * 8;
}
// B-matrix (32x16 bf16) K index: same as A (lanes 0-15: K 0..7,16..23; lanes 16-31: +8)
__device__ __forceinline__ int kmapB(int v, int half) {
    return 2 * v + (v >= 4 ? 8 : 0) + half * 8;
}

// 16-byte async copy global -> LDS (per-lane addresses), tracked by ASYNCcnt.
// Generic pointers to __shared__ carry the LDS byte offset in their low 32
// bits (LDS aperture), which is exactly what the VDST operand wants.
__device__ __forceinline__ void async_ld16(void* lds, const void* g) {
    asm volatile("global_load_async_to_lds_b128 %0, %1, off"
                 :: "v"((unsigned)(unsigned long long)lds),
                    "v"((unsigned long long)g)
                 : "memory");
}
__device__ __forceinline__ void wait_async0() {
    asm volatile("s_wait_asynccnt 0" ::: "memory");
}

// ---------------------------------------------------------------------------
// GEMM: C[M,N] = X[M,1024] * W[N,1024]^T + bias  (M=4096, N=1024)
// MODE 0: X is f32, output bf16 scaled, scattered to [B,H,S,Dh] head layout.
// MODE 1: X is bf16 (async-staged), output f32 row-major -> d_out.
// Block: 256 threads = 8 waves, 128x128 tile; wave = 32(M) x 64(N).
// ---------------------------------------------------------------------------
template <int MODE>
__global__ __launch_bounds__(256) void gemm_kernel(
    const void* __restrict__ Xv, const float* __restrict__ W,
    const float* __restrict__ bias, void* __restrict__ Out, float scale)
{
    __shared__ __align__(16) bf16_t Xs[128][34];  // +2 pad: bank-conflict free
    __shared__ __align__(16) bf16_t Ws[128][34];

    const int tid  = threadIdx.x;
    const int wave = tid >> 5;
    const int lane = tid & 31;
    const int ln15 = lane & 15;
    const int half = lane >> 4;
    const int wm   = wave & 3;   // 4 row-strips of 32
    const int wn   = wave >> 2;  // 2 col-strips of 64
    const int rowBase = blockIdx.y * 128;
    const int colBase = blockIdx.x * 128;

    v8f acc[2][4];
    #pragma unroll
    for (int i = 0; i < 2; i++)
        #pragma unroll
        for (int j = 0; j < 4; j++)
            #pragma unroll
            for (int e = 0; e < 8; e++) acc[i][j][e] = 0.0f;

    for (int k0 = 0; k0 < D_MODEL; k0 += 32) {
        __syncthreads();
        // ---- stage X tile 128x32 ----
        if (MODE == 0) {
            const float* X = (const float*)Xv;
            #pragma unroll
            for (int i = 0; i < 4; i++) {
                int f = tid + 256 * i;          // 1024 float4 units
                int r = f >> 3, c = f & 7;
                const float4 x = *(const float4*)(X + (size_t)(rowBase + r) * D_MODEL + k0 + c * 4);
                bf16_t* d = &Xs[r][c * 4];
                d[0] = (bf16_t)x.x; d[1] = (bf16_t)x.y;
                d[2] = (bf16_t)x.z; d[3] = (bf16_t)x.w;
            }
        } else {
            const bf16_t* X = (const bf16_t*)Xv;
            #pragma unroll
            for (int i = 0; i < 2; i++) {
                int f = tid + 256 * i;          // 512 uint4 units (8 bf16 each)
                int r = f >> 2, c = f & 3;
                async_ld16(&Xs[r][c * 8],
                           X + (size_t)(rowBase + r) * D_MODEL + k0 + c * 8);
            }
        }
        // ---- stage W tile 128x32 (rows of W are output columns) ----
        #pragma unroll
        for (int i = 0; i < 4; i++) {
            int f = tid + 256 * i;
            int r = f >> 3, c = f & 7;
            const float4 x = *(const float4*)(W + (size_t)(colBase + r) * D_MODEL + k0 + c * 4);
            bf16_t* d = &Ws[r][c * 4];
            d[0] = (bf16_t)x.x; d[1] = (bf16_t)x.y;
            d[2] = (bf16_t)x.z; d[3] = (bf16_t)x.w;
        }
        // warm the next k-slice while this one is consumed
        if (k0 + 32 < D_MODEL) {
            __builtin_prefetch(W + (size_t)(colBase + (tid >> 1)) * D_MODEL +
                                   (k0 + 32) + (tid & 1) * 16, 0, 1);
            if (MODE == 0)
                __builtin_prefetch((const float*)Xv +
                                       (size_t)(rowBase + (tid >> 1)) * D_MODEL +
                                       (k0 + 32) + (tid & 1) * 16, 0, 1);
        }
        if (MODE == 1) wait_async0();
        __syncthreads();

        FragB16 afr[2];
        #pragma unroll
        for (int tm = 0; tm < 2; tm++) {
            int row = wm * 32 + tm * 16 + ln15;
            #pragma unroll
            for (int v = 0; v < 8; v++)
                afr[tm].u[v] = *(const unsigned*)&Xs[row][kmapA(v, half)];
        }
        FragB16 bfr[4];
        #pragma unroll
        for (int tn = 0; tn < 4; tn++) {
            int n = wn * 64 + tn * 16 + ln15;
            #pragma unroll
            for (int v = 0; v < 8; v++)
                bfr[tn].u[v] = *(const unsigned*)&Ws[n][kmapB(v, half)];
        }
        #pragma unroll
        for (int tm = 0; tm < 2; tm++)
            #pragma unroll
            for (int tn = 0; tn < 4; tn++)
                acc[tm][tn] = __builtin_amdgcn_wmma_f32_16x16x32_bf16(
                    false, afr[tm].v, false, bfr[tn].v, (short)0, acc[tm][tn],
                    false, false);
    }

    // ---- epilogue ----
    #pragma unroll
    for (int tm = 0; tm < 2; tm++) {
        #pragma unroll
        for (int tn = 0; tn < 4; tn++) {
            const int col  = colBase + wn * 64 + tn * 16 + ln15;
            const float bc = bias[col];
            #pragma unroll
            for (int r = 0; r < 8; r++) {
                const int row = rowBase + wm * 32 + tm * 16 + half * 8 + r;
                const float val = acc[tm][tn][r] + bc;
                if (MODE == 0) {
                    bf16_t* O = (bf16_t*)Out;
                    const int b = row >> 11, s = row & (SEQ - 1);
                    const int h = col >> 6,  dh = col & (D_H - 1);
                    O[((size_t)(b * NUM_HEADS + h) * SEQ + s) * D_H + dh] =
                        (bf16_t)(val * scale);
                } else {
                    float* O = (float*)Out;
                    O[(size_t)row * D_MODEL + col] = val;
                }
            }
        }
    }
}

// ---------------------------------------------------------------------------
// Flash attention: block = 128 query rows x one (b,h); 8 waves, each owns a
// 16-row strip.  KV streamed in 64-wide chunks; K and mask chunks are
// async-copied into LDS, V goes through VGPRs to land transposed.  Q is
// pre-scaled by (1/sqrt(Dh)) * log2(e) so the softmax uses exp2.
// ---------------------------------------------------------------------------
__global__ __launch_bounds__(256) void attn_kernel(
    const bf16_t* __restrict__ Qh, const bf16_t* __restrict__ Kh,
    const bf16_t* __restrict__ Vh, const unsigned char* __restrict__ mask,
    bf16_t* __restrict__ Ah)
{
    __shared__ __align__(16) bf16_t        Ks[64][72];   // K chunk [kv][dh]
    __shared__ __align__(16) bf16_t        Vt[64][72];   // V^T chunk [dh][kv]
    __shared__ __align__(16) unsigned char Ms[128][64];  // mask [q][kv]
    __shared__ __align__(16) bf16_t        Ps[128][72];  // probs [q][kv]

    const int tid  = threadIdx.x;
    const int wave = tid >> 5;
    const int lane = tid & 31;
    const int ln15 = lane & 15;
    const int half = lane >> 4;
    const int bh   = blockIdx.y;           // b*16 + h
    const int b    = bh >> 4;
    const int h    = bh & 15;
    const int q0   = blockIdx.x * 128;

    // Q fragments for this wave's 16 rows (2 frags cover Dh=64)
    FragB16 qfr[2];
    {
        const bf16_t* Qrow = Qh + ((size_t)bh * SEQ + q0 + wave * 16 + ln15) * D_H;
        #pragma unroll
        for (int kc = 0; kc < 2; kc++)
            #pragma unroll
            for (int v = 0; v < 8; v++)
                qfr[kc].u[v] = *(const unsigned*)(Qrow + kc * 32 + kmapA(v, half));
    }

    v8f oacc[4];
    #pragma unroll
    for (int d = 0; d < 4; d++)
        #pragma unroll
        for (int e = 0; e < 8; e++) oacc[d][e] = 0.0f;
    float mrow[8], lrow[8];
    #pragma unroll
    for (int r = 0; r < 8; r++) { mrow[r] = -__builtin_inff(); lrow[r] = 0.0f; }

    const bf16_t* Kbase = Kh + (size_t)bh * SEQ * D_H;
    const bf16_t* Vbase = Vh + (size_t)bh * SEQ * D_H;

    for (int kv0 = 0; kv0 < SEQ; kv0 += 64) {
        __syncthreads();
        // ---- K chunk 64x64 -> LDS, async (8KB contiguous) ----
        #pragma unroll
        for (int i = 0; i < 2; i++) {
            int f = tid + 256 * i;             // 512 uint4 units
            int r = f >> 3, c = f & 7;
            async_ld16(&Ks[r][c * 8], Kbase + (size_t)(kv0 + r) * D_H + c * 8);
        }
        // ---- mask chunk 128x64 bytes, async ----
        #pragma unroll
        for (int i = 0; i < 2; i++) {
            int f = tid + 256 * i;             // 512 uint4 units
            int r = f >> 2, c = f & 3;
            async_ld16(&Ms[r][c * 16],
                       mask + ((size_t)b * SEQ + q0 + r) * SEQ + kv0 + c * 16);
        }
        // ---- V chunk 64x64 -> LDS transposed (through VGPRs) ----
        #pragma unroll
        for (int i = 0; i < 2; i++) {
            int f = tid + 256 * i;
            int r = f >> 3, c = f & 7;         // r = kv row, c*8 = dh base
            uint4 x = *(const uint4*)(Vbase + (size_t)(kv0 + r) * D_H + c * 8);
            const bf16_t* px = (const bf16_t*)&x;
            #pragma unroll
            for (int j = 0; j < 8; j++) Vt[c * 8 + j][r] = px[j];
        }
        // warm next K/V chunk (8KB each) while this one is consumed
        if (kv0 + 64 < SEQ) {
            __builtin_prefetch(Kbase + (size_t)(kv0 + 64) * D_H + tid * 16, 0, 1);
            __builtin_prefetch(Vbase + (size_t)(kv0 + 64) * D_H + tid * 16, 0, 1);
        }
        wait_async0();
        __syncthreads();

        // ---- scores: S = Q (16xDh) * K^T (Dhx64), 4 tiles of 16x16 ----
        v8f sacc[4];
        #pragma unroll
        for (int tn = 0; tn < 4; tn++)
            #pragma unroll
            for (int e = 0; e < 8; e++) sacc[tn][e] = 0.0f;
        #pragma unroll
        for (int tn = 0; tn < 4; tn++) {
            #pragma unroll
            for (int kc = 0; kc < 2; kc++) {
                FragB16 bf;  // B = K^T : B[dh][kv] = Ks[kv][dh]
                #pragma unroll
                for (int v = 0; v < 8; v++)
                    bf.u[v] = *(const unsigned*)&Ks[tn * 16 + ln15][kc * 32 + kmapB(v, half)];
                sacc[tn] = __builtin_amdgcn_wmma_f32_16x16x32_bf16(
                    false, qfr[kc].v, false, bf.v, (short)0, sacc[tn], false, false);
            }
        }

        // ---- mask + online softmax (per 16-lane half: rows half*8+r) ----
        const int qrl_base = wave * 16 + half * 8;
        #pragma unroll
        for (int r = 0; r < 8; r++) {
            const int qrl = qrl_base + r;
            float s[4];
            float rmax = -__builtin_inff();
            #pragma unroll
            for (int tn = 0; tn < 4; tn++) {
                float x = sacc[tn][r];
                if (!Ms[qrl][tn * 16 + ln15]) x = -__builtin_inff();
                s[tn] = x;
                rmax = fmaxf(rmax, x);
            }
            #pragma unroll
            for (int mm = 8; mm >= 1; mm >>= 1)
                rmax = fmaxf(rmax, __shfl_xor(rmax, mm, 32));
            const float newm  = fmaxf(mrow[r], rmax);
            const float alpha = (mrow[r] == -__builtin_inff()) ? 0.0f
                                                               : exp2f(mrow[r] - newm);
            float rsum = 0.0f;
            #pragma unroll
            for (int tn = 0; tn < 4; tn++) {
                float e = (s[tn] == -__builtin_inff()) ? 0.0f : exp2f(s[tn] - newm);
                sacc[tn][r] = e;
                rsum += e;
            }
            #pragma unroll
            for (int mm = 8; mm >= 1; mm >>= 1)
                rsum += __shfl_xor(rsum, mm, 32);
            lrow[r] = lrow[r] * alpha + rsum;
            mrow[r] = newm;
            #pragma unroll
            for (int d = 0; d < 4; d++) oacc[d][r] *= alpha;
            #pragma unroll
            for (int tn = 0; tn < 4; tn++)
                Ps[qrl][tn * 16 + ln15] = (bf16_t)sacc[tn][r];
        }
        __syncthreads();

        // ---- O += P (16x64) * V (64x64) ----
        #pragma unroll
        for (int kc = 0; kc < 2; kc++) {
            FragB16 pa;  // A = probs, re-read in A layout from LDS
            #pragma unroll
            for (int v = 0; v < 8; v++)
                pa.u[v] = *(const unsigned*)&Ps[wave * 16 + ln15][kc * 32 + kmapA(v, half)];
            #pragma unroll
            for (int dt = 0; dt < 4; dt++) {
                FragB16 bf;  // B = V : B[kv][dh] = Vt[dh][kv]
                #pragma unroll
                for (int v = 0; v < 8; v++)
                    bf.u[v] = *(const unsigned*)&Vt[dt * 16 + ln15][kc * 32 + kmapB(v, half)];
                oacc[dt] = __builtin_amdgcn_wmma_f32_16x16x32_bf16(
                    false, pa.v, false, bf.v, (short)0, oacc[dt], false, false);
            }
        }
    }

    // ---- normalize and store A in token-major [B,S,H*Dh] bf16 ----
    const int srow_base = q0 + wave * 16 + half * 8;
    #pragma unroll
    for (int r = 0; r < 8; r++) {
        const float inv  = 1.0f / fmaxf(lrow[r], 1e-30f);
        const int   srow = srow_base + r;
        const size_t off = ((size_t)b * SEQ + srow) * D_MODEL + h * D_H;
        #pragma unroll
        for (int d = 0; d < 4; d++)
            Ah[off + d * 16 + ln15] = (bf16_t)(oacc[d][r] * inv);
    }
}

// ---------------------------------------------------------------------------
extern "C" void kernel_launch(void* const* d_in, const int* in_sizes, int n_in,
                              void* d_out, int out_size, void* d_ws, size_t ws_size,
                              hipStream_t stream)
{
    const float*         q    = (const float*)d_in[0];
    const float*         k    = (const float*)d_in[1];
    const float*         v    = (const float*)d_in[2];
    const unsigned char* mask = (const unsigned char*)d_in[3];  // bool tensor
    const float* Wq = (const float*)d_in[4];  const float* bq = (const float*)d_in[5];
    const float* Wk = (const float*)d_in[6];  const float* bk = (const float*)d_in[7];
    const float* Wv = (const float*)d_in[8];  const float* bv = (const float*)d_in[9];
    const float* Wo = (const float*)d_in[10]; const float* bo = (const float*)d_in[11];

    // workspace layout: bf16 Qh/Kh/Vh [B,H,S,Dh] + bf16 A [B,S,D] = 32 MB
    const size_t one = (size_t)BATCH * NUM_HEADS * SEQ * D_H * sizeof(bf16_t);  // 8 MB
    char* ws = (char*)d_ws;
    bf16_t* Qh = (bf16_t*)(ws);
    bf16_t* Kh = (bf16_t*)(ws + one);
    bf16_t* Vh = (bf16_t*)(ws + 2 * one);
    bf16_t* Ah = (bf16_t*)(ws + 3 * one);

    const dim3 blk(256);
    const dim3 ggemm(D_MODEL / 128, (BATCH * SEQ) / 128);  // (8, 32)
    const float qscale = 0.125f * 1.44269504088896340736f; // 1/sqrt(64) * log2(e)

    gemm_kernel<0><<<ggemm, blk, 0, stream>>>(q, Wq, bq, Qh, qscale);
    gemm_kernel<0><<<ggemm, blk, 0, stream>>>(k, Wk, bk, Kh, 1.0f);
    gemm_kernel<0><<<ggemm, blk, 0, stream>>>(v, Wv, bv, Vh, 1.0f);

    const dim3 gattn(SEQ / 128, BATCH * NUM_HEADS);        // (16, 32)
    attn_kernel<<<gattn, blk, 0, stream>>>(Qh, Kh, Vh, mask, Ah);

    gemm_kernel<1><<<ggemm, blk, 0, stream>>>(Ah, Wo, bo, d_out, 1.0f);
}
